// MultiHeadAttention_14173392076997
// MI455X (gfx1250) — matrix operand order repaired and checked by reading, not compiled, not run on hardware
//
#include <hip/hip_runtime.h>
#include <hip/hip_bf16.h>
#include <math.h>

// ---------------------------------------------------------------------------
// MI455X / gfx1250 multi-head attention, bf16 WMMA + f32 accumulation.
// B=2, T=2048, C=1024, H=16, d=64. Flash-style fused softmax (no TxT
// materialization: that would be 536MB of HBM traffic vs ~50MB mandatory).
// ---------------------------------------------------------------------------

typedef __bf16 bf16_t;
typedef __attribute__((ext_vector_type(16))) __bf16 v16bf;
typedef __attribute__((ext_vector_type(8)))  __bf16 v8bf;
typedef __attribute__((ext_vector_type(8)))  float  v8f;

#define WMMA_BF16(a, b, c) \
  __builtin_amdgcn_wmma_f32_16x16x32_bf16(false, (a), false, (b), (short)0, (c), false, false)

static constexpr int Bsz = 2;
static constexpr int T   = 2048;
static constexpr int C   = 1024;
static constexpr int H   = 16;
static constexpr int D   = 64;
static constexpr int M   = Bsz * T;      // 4096 rows (tokens)
static constexpr int N_QKV = 3 * C;      // 3072

// ---- fragment loaders (layouts per CDNA5 ISA 7.12.2) ----------------------
// A (16x32 bf16, MxK): lane L -> row M = L%16, half h = L/16.
//   elements e=0..7  : K = h*8 + e          (contiguous 16B)
//   elements e=8..15 : K = 16 + h*8 + (e-8) (contiguous 16B)
static __device__ __forceinline__
v16bf load_a_frag(const bf16_t* __restrict__ src, int ld, int row_base, int k_base, int lane) {
  const int m = lane & 15;
  const int h = lane >> 4;
  const bf16_t* p = src + (size_t)(row_base + m) * ld + k_base + h * 8;
  v8bf lo = *(const v8bf*)(p);
  v8bf hi = *(const v8bf*)(p + 16);
  v16bf a;
#pragma unroll
  for (int i = 0; i < 8; ++i) { a[i] = lo[i]; a[8 + i] = hi[i]; }
  return a;
}

// B (32x16 bf16, KxN), source stored row-major [N, K] (output-feature major):
// lane L -> col N = L%16, half h = L/16, elements e=0..7: K = h*8 + e,
// e=8..15: K = 16 + h*8 + (e-8) => two contiguous 16-byte runs per lane.
static __device__ __forceinline__
v16bf load_b_frag(const bf16_t* __restrict__ src, int ld, int n_base, int k_base, int lane) {
  const int n = lane & 15;
  const int h = lane >> 4;
  const bf16_t* p = src + (size_t)(n_base + n) * ld + k_base + h * 8;
  v8bf lo = *(const v8bf*)(p);
  v8bf hi = *(const v8bf*)(p + 16);
  v16bf b;
#pragma unroll
  for (int i = 0; i < 8; ++i) { b[i] = lo[i]; b[8 + i] = hi[i]; }
  return b;
}

// ---- fp32 -> bf16 bulk convert (vectorized, bandwidth-bound) --------------
__global__ __launch_bounds__(256)
void cvt_f32_bf16(const float* __restrict__ src, bf16_t* __restrict__ dst, int n8) {
  int i = blockIdx.x * blockDim.x + threadIdx.x;
  for (; i < n8; i += gridDim.x * blockDim.x) {
    const float* s = src + (size_t)i * 8;
    v8bf o;
#pragma unroll
    for (int j = 0; j < 8; ++j) o[j] = (bf16_t)s[j];
    *(v8bf*)(dst + (size_t)i * 8) = o;
  }
}

// ---- Kernel 1: fused QKV projection, M=4096 N=3072 K=1024 -----------------
// Wave tile 64x64 (16 f32 accumulator frags, 16 WMMAs per K=32 step).
// Block = 4 waves covering 64 rows x 256 cols. Scatters Q,K -> [B,H,T,d],
// V -> transposed [B,H,d,T] so attention B-fragments are contiguous.
__global__ __launch_bounds__(128)
void qkv_gemm(const bf16_t* __restrict__ xb, const bf16_t* __restrict__ wab,
              bf16_t* __restrict__ Qm, bf16_t* __restrict__ Km, bf16_t* __restrict__ Vt) {
  const int lane  = threadIdx.x & 31;
  const int wave  = threadIdx.x >> 5;               // 0..3
  const int mBase = blockIdx.x * 64;                // [0,4096)
  const int nBase = blockIdx.y * 256 + wave * 64;   // [0,3072)

  v8f acc[4][4] = {};
  for (int k = 0; k < C; k += 32) {
    v16bf a[4], b[4];
#pragma unroll
    for (int mi = 0; mi < 4; ++mi) a[mi] = load_a_frag(xb, C, mBase + mi * 16, k, lane);
#pragma unroll
    for (int ni = 0; ni < 4; ++ni) b[ni] = load_b_frag(wab, C, nBase + ni * 16, k, lane);
    if (k + 32 < C) {  // prefetch next K slab
      __builtin_prefetch(xb + (size_t)(mBase + (lane & 15)) * C + k + 32, 0, 1);
      __builtin_prefetch(wab + (size_t)(nBase + (lane & 15)) * C + k + 32, 0, 1);
    }
#pragma unroll
    for (int mi = 0; mi < 4; ++mi)
#pragma unroll
      for (int ni = 0; ni < 4; ++ni)
        acc[mi][ni] = WMMA_BF16(a[mi], b[ni], acc[mi][ni]);
  }

  // D-frag: lane -> (row = r + 8*(lane/16), col = lane%16)
  const int hlf = lane >> 4, nn = lane & 15;
#pragma unroll
  for (int mi = 0; mi < 4; ++mi)
#pragma unroll
    for (int ni = 0; ni < 4; ++ni)
#pragma unroll
      for (int r = 0; r < 8; ++r) {
        const int row = mBase + mi * 16 + r + 8 * hlf;   // token index
        const int f   = nBase + ni * 16 + nn;            // fused feature
        const int bidx = row >> 11, t = row & (T - 1);
        const int which = f >> 10, c = f & (C - 1);
        const int hh = c >> 6, dd = c & (D - 1);
        const bf16_t v = (bf16_t)acc[mi][ni][r];
        const size_t head = (size_t)bidx * H + hh;
        if (which == 0)      Qm[(head * T + t) * D + dd] = v;
        else if (which == 1) Km[(head * T + t) * D + dd] = v;
        else                 Vt[(head * D + dd) * T + t] = v;  // V transposed
      }
}

// ---- Kernel 2: fused flash attention --------------------------------------
// grid = (T/64, B*H), block = 128 (4 waves). Each wave: 16 query rows,
// streams 32-wide key tiles, online softmax, P staged via per-wave LDS slab
// to re-layout D-frag -> A-frag, then P(16x32) x V(32x64) into 4 accumulators.
__global__ __launch_bounds__(128)
void flash_attn(const bf16_t* __restrict__ Qm, const bf16_t* __restrict__ Km,
                const bf16_t* __restrict__ Vt, bf16_t* __restrict__ y) {
  __shared__ bf16_t pstage[4][16 * 32];

  const int lane = threadIdx.x & 31;
  const int wave = threadIdx.x >> 5;
  const int bh   = blockIdx.y;                 // b*H + h
  const int b    = bh >> 4, hh = bh & 15;
  const int qBase = blockIdx.x * 64 + wave * 16;

  const bf16_t* Qp = Qm + (size_t)bh * T * D;
  const bf16_t* Kp = Km + (size_t)bh * T * D;
  const bf16_t* Vp = Vt + (size_t)bh * D * T;  // rows = head-dim, cols = T

  v16bf aQ0 = load_a_frag(Qp, D, qBase, 0,  lane);
  v16bf aQ1 = load_a_frag(Qp, D, qBase, 32, lane);

  float mrow[8], lrow[8];
  v8f acc[4] = {};
#pragma unroll
  for (int r = 0; r < 8; ++r) { mrow[r] = -3.402823466e38f; lrow[r] = 0.f; }
  const float scale = 0.125f;                  // 1/sqrt(64)

  const int hlf = lane >> 4, nn = lane & 15;
  bf16_t* ps = &pstage[wave][0];

  for (int kt = 0; kt < T; kt += 32) {
    // --- S = Q K^T for a 16x32 score tile (two 16x16 D-frags) ---
    v8f s[2] = {};
#pragma unroll
    for (int ng = 0; ng < 2; ++ng) {
      v16bf bk0 = load_b_frag(Kp, D, kt + ng * 16, 0,  lane);
      v16bf bk1 = load_b_frag(Kp, D, kt + ng * 16, 32, lane);
      s[ng] = WMMA_BF16(aQ0, bk0, s[ng]);
      s[ng] = WMMA_BF16(aQ1, bk1, s[ng]);
    }

    // --- online softmax: row max across the 16 lanes of each half ---
    float alpha[8];
#pragma unroll
    for (int r = 0; r < 8; ++r) {
      float v = fmaxf(s[0][r], s[1][r]);
      v = fmaxf(v, __shfl_xor(v, 1));
      v = fmaxf(v, __shfl_xor(v, 2));
      v = fmaxf(v, __shfl_xor(v, 4));
      v = fmaxf(v, __shfl_xor(v, 8));
      const float mn = fmaxf(mrow[r], v * scale);
      alpha[r] = __expf(mrow[r] - mn);
      mrow[r] = mn;
    }
    // p = exp(scale*s - m), row sums, rescale running state
#pragma unroll
    for (int r = 0; r < 8; ++r) {
      s[0][r] = __expf(s[0][r] * scale - mrow[r]);
      s[1][r] = __expf(s[1][r] * scale - mrow[r]);
      float sum = s[0][r] + s[1][r];
      sum += __shfl_xor(sum, 1);
      sum += __shfl_xor(sum, 2);
      sum += __shfl_xor(sum, 4);
      sum += __shfl_xor(sum, 8);
      lrow[r] = lrow[r] * alpha[r] + sum;
    }
#pragma unroll
    for (int j = 0; j < 4; ++j)
#pragma unroll
      for (int r = 0; r < 8; ++r) acc[j][r] *= alpha[r];

    // --- stage P (D-frag layout) -> LDS 16x32 row-major bf16 ---
#pragma unroll
    for (int ng = 0; ng < 2; ++ng)
#pragma unroll
      for (int r = 0; r < 8; ++r)
        ps[(r + 8 * hlf) * 32 + ng * 16 + nn] = (bf16_t)s[ng][r];
    // intra-wave LDS RAW: hardware keeps DS ops in order per wave; the wait +
    // memory clobber stops the compiler hoisting the reads above the writes.
    __asm__ volatile("s_wait_dscnt 0" ::: "memory");

    // --- reload P as a 16x32 A-fragment ---
    v16bf aP;
    {
      const bf16_t* pr = ps + (lane & 15) * 32 + hlf * 8;
      v8bf lo = *(const v8bf*)(pr);
      v8bf hi = *(const v8bf*)(pr + 16);
#pragma unroll
      for (int i = 0; i < 8; ++i) { aP[i] = lo[i]; aP[8 + i] = hi[i]; }
    }

    // --- O += P x V  (V^T rows = head-dim => contiguous B-frag loads) ---
#pragma unroll
    for (int ndg = 0; ndg < 4; ++ndg) {
      v16bf bv = load_b_frag(Vp, T, ndg * 16, kt, lane);
      acc[ndg] = WMMA_BF16(aP, bv, acc[ndg]);
    }
  }

  // --- epilogue: normalize by l, write y[b,t,c] as bf16 for the out-proj ---
#pragma unroll
  for (int ndg = 0; ndg < 4; ++ndg)
#pragma unroll
    for (int r = 0; r < 8; ++r) {
      const int q = qBase + r + 8 * hlf;
      const int c = hh * D + ndg * 16 + nn;
      y[((size_t)b * T + q) * C + c] = (bf16_t)(acc[ndg][r] / lrow[r]);
    }
}

// ---- Kernel 3: output projection, M=4096 N=1024 K=1024, + bias, f32 out ---
__global__ __launch_bounds__(128)
void proj_gemm(const bf16_t* __restrict__ yb, const bf16_t* __restrict__ wpb,
               const float* __restrict__ bias, float* __restrict__ out) {
  const int lane  = threadIdx.x & 31;
  const int wave  = threadIdx.x >> 5;
  const int mBase = blockIdx.x * 64;
  const int nBase = blockIdx.y * 256 + wave * 64;

  v8f acc[4][4] = {};
  for (int k = 0; k < C; k += 32) {
    v16bf a[4], bfr[4];
#pragma unroll
    for (int mi = 0; mi < 4; ++mi) a[mi] = load_a_frag(yb, C, mBase + mi * 16, k, lane);
#pragma unroll
    for (int ni = 0; ni < 4; ++ni) bfr[ni] = load_b_frag(wpb, C, nBase + ni * 16, k, lane);
    if (k + 32 < C) {
      __builtin_prefetch(yb + (size_t)(mBase + (lane & 15)) * C + k + 32, 0, 1);
      __builtin_prefetch(wpb + (size_t)(nBase + (lane & 15)) * C + k + 32, 0, 1);
    }
#pragma unroll
    for (int mi = 0; mi < 4; ++mi)
#pragma unroll
      for (int ni = 0; ni < 4; ++ni)
        acc[mi][ni] = WMMA_BF16(a[mi], bfr[ni], acc[mi][ni]);
  }

  const int hlf = lane >> 4, nn = lane & 15;
#pragma unroll
  for (int mi = 0; mi < 4; ++mi)
#pragma unroll
    for (int ni = 0; ni < 4; ++ni) {
      const int o = nBase + ni * 16 + nn;
      const float bo = bias[o];
#pragma unroll
      for (int r = 0; r < 8; ++r) {
        const int row = mBase + mi * 16 + r + 8 * hlf;
        out[(size_t)row * C + o] = acc[mi][ni][r] + bo;
      }
    }
}

// ---------------------------------------------------------------------------
extern "C" void kernel_launch(void* const* d_in, const int* in_sizes, int n_in,
                              void* d_out, int out_size, void* d_ws, size_t ws_size,
                              hipStream_t stream) {
  const float* x      = (const float*)d_in[0];   // [B,T,C]   = 4,194,304
  const float* W_attn = (const float*)d_in[1];   // [3C,C]    = 3,145,728
  const float* W_proj = (const float*)d_in[2];   // [C,C]     = 1,048,576
  const float* b_proj = (const float*)d_in[3];   // [C]
  float* out = (float*)d_out;

  char* w = (char*)d_ws;
  bf16_t* xb  = (bf16_t*)(w + (size_t)0);             // 8 MB
  bf16_t* wab = (bf16_t*)(w + ((size_t)8  << 20));    // 6 MB
  bf16_t* wpb = (bf16_t*)(w + ((size_t)14 << 20));    // 2 MB
  bf16_t* Qm  = (bf16_t*)(w + ((size_t)16 << 20));    // 8 MB [B,H,T,d]
  bf16_t* Km  = (bf16_t*)(w + ((size_t)24 << 20));    // 8 MB [B,H,T,d]
  bf16_t* Vt  = (bf16_t*)(w + ((size_t)32 << 20));    // 8 MB [B,H,d,T]
  bf16_t* yb  = (bf16_t*)(w + ((size_t)40 << 20));    // 8 MB [B,T,C]

  // 0) fp32 -> bf16 converts (bandwidth-bound, ~48MB total traffic)
  cvt_f32_bf16<<<512, 256, 0, stream>>>(x,      xb,  (M * C) / 8);
  cvt_f32_bf16<<<512, 256, 0, stream>>>(W_attn, wab, (N_QKV * C) / 8);
  cvt_f32_bf16<<<256, 256, 0, stream>>>(W_proj, wpb, (C * C) / 8);

  // 1) fused QKV projection: 4096x3072x1024
  qkv_gemm<<<dim3(M / 64, N_QKV / 256), 128, 0, stream>>>(xb, wab, Qm, Km, Vt);

  // 2) fused flash attention: grid (T/64, B*H)
  flash_attn<<<dim3(T / 64, Bsz * H), 128, 0, stream>>>(Qm, Km, Vt, yb);

  // 3) output projection + bias: 4096x1024x1024, fp32 out
  proj_gemm<<<dim3(M / 64, C / 256), 128, 0, stream>>>(yb, wpb, b_proj, out);

  (void)in_sizes; (void)n_in; (void)out_size; (void)ws_size;
}
